// EglGATConv_52664888984197
// MI455X (gfx1250) — matrix operand order repaired and checked by reading, not compiled, not run on hardware
//
#include <hip/hip_runtime.h>
#include <hip/hip_bf16.h>

#define NN 50000
#define NE 800000
#define INF 128
#define HEADS 4
#define DIMS 32
#define OUTF 128   // HEADS*DIMS
#define NEG_SLOPE 0.2f

typedef __attribute__((ext_vector_type(16))) __bf16 v16bf;
typedef __attribute__((ext_vector_type(8)))  float  v8f;

// ---------------------------------------------------------------------------
// Fused GEMM:  h = feat @ fc_w^T   (fused cols 0..127  -> hbuf)
//              res = feat @ res_w^T (fused cols 128..255 -> out, initializes it)
// grid.x = NN/16 M-tiles, block = 256 threads (8 waves), wave w owns col-tiles
// {2w, 2w+1} of the 256-wide fused output.  bf16 WMMA, f32 accumulate.
// ---------------------------------------------------------------------------
__global__ __launch_bounds__(256) void gat_gemm(const float* __restrict__ feat,
                                                const float* __restrict__ fc_w,
                                                const float* __restrict__ res_w,
                                                float* __restrict__ hbuf,
                                                float* __restrict__ out)
{
    const int wave = threadIdx.x >> 5;      // 0..7
    const int lane = threadIdx.x & 31;
    const int half = lane >> 4;             // 0 / 1
    const int r    = lane & 15;
    const int mtile = blockIdx.x;
    const int m = mtile * 16 + r;           // A-fragment row for this lane

    const float* arow = feat + (size_t)m * INF;

    const int nt0 = wave * 2;
    const int nt1 = wave * 2 + 1;
    const int col0 = nt0 * 16 + r;          // fused output column (B col) per lane
    const int col1 = nt1 * 16 + r;
    const float* wrow0 = (col0 < OUTF) ? (fc_w  + (size_t)col0          * INF)
                                       : (res_w + (size_t)(col0 - OUTF) * INF);
    const float* wrow1 = (col1 < OUTF) ? (fc_w  + (size_t)col1          * INF)
                                       : (res_w + (size_t)(col1 - OUTF) * INF);

    v8f acc0 = {}; v8f acc1 = {};

    auto loadB = [&](const float* wrow, int k0) -> v16bf {
        // B(32x16) 16-bit layout: lane half 0 -> K {k0+0..7, k0+16..23}, lane half 1 -> K {k0+8..15, k0+24..31}
        const float4* p = (const float4*)(wrow + k0 + half * 8);
        const float4* q = (const float4*)(wrow + k0 + 16 + half * 8);
        float4 g0 = p[0], g1 = p[1], g2 = q[0], g3 = q[1];
        v16bf b;
        b[0]=(__bf16)g0.x;  b[1]=(__bf16)g0.y;  b[2]=(__bf16)g0.z;  b[3]=(__bf16)g0.w;
        b[4]=(__bf16)g1.x;  b[5]=(__bf16)g1.y;  b[6]=(__bf16)g1.z;  b[7]=(__bf16)g1.w;
        b[8]=(__bf16)g2.x;  b[9]=(__bf16)g2.y;  b[10]=(__bf16)g2.z; b[11]=(__bf16)g2.w;
        b[12]=(__bf16)g3.x; b[13]=(__bf16)g3.y; b[14]=(__bf16)g3.z; b[15]=(__bf16)g3.w;
        return b;
    };

    #pragma unroll
    for (int k0 = 0; k0 < INF; k0 += 32) {
        // A(16x32) 16-bit layout: lane half 0 -> K {k0+0..7, k0+16..23},
        //                         lane half 1 -> K {k0+8..15, k0+24..31}
        const float4* p0 = (const float4*)(arow + k0 + half * 8);
        const float4* p1 = (const float4*)(arow + k0 + 16 + half * 8);
        float4 f0 = p0[0], f1 = p0[1], f2 = p1[0], f3 = p1[1];
        v16bf a;
        a[0]=(__bf16)f0.x;  a[1]=(__bf16)f0.y;  a[2]=(__bf16)f0.z;  a[3]=(__bf16)f0.w;
        a[4]=(__bf16)f1.x;  a[5]=(__bf16)f1.y;  a[6]=(__bf16)f1.z;  a[7]=(__bf16)f1.w;
        a[8]=(__bf16)f2.x;  a[9]=(__bf16)f2.y;  a[10]=(__bf16)f2.z; a[11]=(__bf16)f2.w;
        a[12]=(__bf16)f3.x; a[13]=(__bf16)f3.y; a[14]=(__bf16)f3.z; a[15]=(__bf16)f3.w;

        v16bf b0 = loadB(wrow0, k0);
        v16bf b1 = loadB(wrow1, k0);

        acc0 = __builtin_amdgcn_wmma_f32_16x16x32_bf16(false, a, false, b0,
                                                       (short)0, acc0, false, false);
        acc1 = __builtin_amdgcn_wmma_f32_16x16x32_bf16(false, a, false, b1,
                                                       (short)0, acc1, false, false);
    }

    // C/D layout: VGPR vi, lanes 0-15 -> row vi, lanes 16-31 -> row vi+8; col = r
    #pragma unroll
    for (int vi = 0; vi < 8; ++vi) {
        const int row = mtile * 16 + vi + half * 8;
        const int c0 = nt0 * 16 + r;
        const int c1 = nt1 * 16 + r;
        if (c0 < OUTF) hbuf[(size_t)row * OUTF + c0] = acc0[vi];
        else           out [(size_t)row * OUTF + (c0 - OUTF)] = acc0[vi];
        if (c1 < OUTF) hbuf[(size_t)row * OUTF + c1] = acc1[vi];
        else           out [(size_t)row * OUTF + (c1 - OUTF)] = acc1[vi];
    }
}

// ---------------------------------------------------------------------------
// el[n,h] = dot(h[n,h,:], attn_l[h,:]) ; er likewise. One thread per (n,h).
// ---------------------------------------------------------------------------
__global__ __launch_bounds__(256) void gat_logits(const float* __restrict__ hbuf,
                                                  const float* __restrict__ attn_l,
                                                  const float* __restrict__ attn_r,
                                                  float* __restrict__ el,
                                                  float* __restrict__ er)
{
    const int t = blockIdx.x * blockDim.x + threadIdx.x;
    if (t >= NN * HEADS) return;
    const int n  = t >> 2;
    const int hh = t & 3;
    const float4* hv = (const float4*)(hbuf + (size_t)n * OUTF + hh * DIMS);
    const float4* al = (const float4*)(attn_l + hh * DIMS);
    const float4* ar = (const float4*)(attn_r + hh * DIMS);
    float sl = 0.f, sr = 0.f;
    #pragma unroll
    for (int j = 0; j < DIMS / 4; ++j) {
        float4 h4 = hv[j], a4 = al[j], b4 = ar[j];
        sl += h4.x * a4.x + h4.y * a4.y + h4.z * a4.z + h4.w * a4.w;
        sr += h4.x * b4.x + h4.y * b4.y + h4.z * b4.z + h4.w * b4.w;
    }
    el[t] = sl;
    er[t] = sr;
}

// ---------------------------------------------------------------------------
// Per-edge coefficients + softmax denominators. One thread per edge (H=4
// heads contiguous -> float4 load/store + 4 f32 atomics).
// ---------------------------------------------------------------------------
__device__ __forceinline__ float lrelu_exp(float x) {
    float y = (x > 0.f) ? x : NEG_SLOPE * x;
    return __expf(y);
}

__global__ __launch_bounds__(256) void gat_edge_coeff(const int* __restrict__ src,
                                                      const int* __restrict__ dst,
                                                      const float* __restrict__ el,
                                                      const float* __restrict__ er,
                                                      float* __restrict__ coeff,
                                                      float* __restrict__ denom)
{
    const int e = blockIdx.x * blockDim.x + threadIdx.x;
    if (e >= NE) return;
    const int s = src[e];
    const int d = dst[e];
    float4 a = *(const float4*)(el + (size_t)s * HEADS);
    float4 b = *(const float4*)(er + (size_t)d * HEADS);
    float4 c;
    c.x = lrelu_exp(a.x + b.x);
    c.y = lrelu_exp(a.y + b.y);
    c.z = lrelu_exp(a.z + b.z);
    c.w = lrelu_exp(a.w + b.w);
    *(float4*)(coeff + (size_t)e * HEADS) = c;
    float* dn = denom + (size_t)d * HEADS;
    atomicAdd(dn + 0, c.x);
    atomicAdd(dn + 1, c.y);
    atomicAdd(dn + 2, c.z);
    atomicAdd(dn + 3, c.w);
}

// ---------------------------------------------------------------------------
// Weighted aggregation: one wave32 per edge, lane = D index, loop over heads.
// out[dst,h,d] += (coeff/denom[dst]) * h[src,h,d]  (f32 global atomics, L2-hot)
// ---------------------------------------------------------------------------
__global__ __launch_bounds__(256) void gat_edge_agg(const int* __restrict__ src,
                                                    const int* __restrict__ dst,
                                                    const float* __restrict__ coeff,
                                                    const float* __restrict__ denom,
                                                    const float* __restrict__ hbuf,
                                                    float* __restrict__ out)
{
    const int gt   = blockIdx.x * blockDim.x + threadIdx.x;
    const int e    = gt >> 5;
    const int lane = gt & 31;
    if (e >= NE) return;
    const int s = src[e];
    const int d = dst[e];
    float4 c  = *(const float4*)(coeff + (size_t)e * HEADS);
    float4 dn = *(const float4*)(denom + (size_t)d * HEADS);
    const float alpha[HEADS] = { c.x / dn.x, c.y / dn.y, c.z / dn.z, c.w / dn.w };
    const float* hrow = hbuf + (size_t)s * OUTF;
    float*       orow = out  + (size_t)d * OUTF;
    #pragma unroll
    for (int hh = 0; hh < HEADS; ++hh) {
        atomicAdd(orow + hh * DIMS + lane, alpha[hh] * hrow[hh * DIMS + lane]);
    }
}

// ---------------------------------------------------------------------------
extern "C" void kernel_launch(void* const* d_in, const int* in_sizes, int n_in,
                              void* d_out, int out_size, void* d_ws, size_t ws_size,
                              hipStream_t stream)
{
    const float* feat   = (const float*)d_in[0];
    const int*   src    = (const int*)  d_in[1];
    const int*   dst    = (const int*)  d_in[2];
    const float* fc_w   = (const float*)d_in[3];
    const float* attn_l = (const float*)d_in[4];
    const float* attn_r = (const float*)d_in[5];
    const float* res_w  = (const float*)d_in[6];
    float* out = (float*)d_out;

    char* ws = (char*)d_ws;
    float* hbuf  = (float*)ws; ws += (size_t)NN * OUTF  * sizeof(float); // 25.6 MB
    float* el    = (float*)ws; ws += (size_t)NN * HEADS * sizeof(float);
    float* er    = (float*)ws; ws += (size_t)NN * HEADS * sizeof(float);
    float* denom = (float*)ws; ws += (size_t)NN * HEADS * sizeof(float);
    float* coeff = (float*)ws; ws += (size_t)NE * HEADS * sizeof(float); // 12.8 MB

    hipMemsetAsync(denom, 0, (size_t)NN * HEADS * sizeof(float), stream);

    gat_gemm<<<NN / 16, 256, 0, stream>>>(feat, fc_w, res_w, hbuf, out);
    gat_logits<<<(NN * HEADS + 255) / 256, 256, 0, stream>>>(hbuf, attn_l, attn_r, el, er);
    gat_edge_coeff<<<(NE + 255) / 256, 256, 0, stream>>>(src, dst, el, er, coeff, denom);
    gat_edge_agg<<<(NE * 32) / 256, 256, 0, stream>>>(src, dst, coeff, denom, hbuf, out);
}
